// GCN_936302871238
// MI455X (gfx1250) — matrix operand order repaired and checked by reading, not compiled, not run on hardware
//
#include <hip/hip_runtime.h>

typedef __attribute__((ext_vector_type(16))) __bf16 v16bf;
typedef __attribute__((ext_vector_type(8)))  float  v8f;

union BF16Frag { unsigned int u[8]; v16bf v; };

__device__ __forceinline__ unsigned short f32_to_bf16(float f) {
    union { float f; unsigned int u; } x; x.f = f;
    unsigned int u = x.u;
    u += 0x7FFFu + ((u >> 16) & 1u);   // round to nearest even
    return (unsigned short)(u >> 16);
}
__device__ __forceinline__ float bf16_to_f32(unsigned short h) {
    union { unsigned int u; float f; } x; x.u = ((unsigned int)h) << 16;
    return x.f;
}

// ---------------------------------------------------------------------------
// Tiled GEMM: C[M x Nout] = A[M x K] @ W[K x Nout], A bf16 (rows padded to
// 128-multiple with zeros), Wt bf16 pre-transposed [Nout x K].
// Output: bf16 (Cbf) or fp32+bias (Cf). Block tile 128x128, K-step 32,
// 8 waves x (2x4) v_wmma_f32_16x16x32_bf16.
// ---------------------------------------------------------------------------
#define TILE_K 32
#define APITCH 40
#define BPITCH 40

__global__ __launch_bounds__(256) void gemm_wmma_bf16(
    const unsigned short* __restrict__ A,    // [Mpad x K] bf16, zero-padded rows
    const unsigned short* __restrict__ Wt,   // [Nout x K] bf16 (transposed)
    const float* __restrict__ bias,          // used in fp32 mode (may be null)
    unsigned short* __restrict__ Cbf,        // bf16 out (null -> fp32 mode)
    float* __restrict__ Cf,                  // fp32 out
    int M, int K, int Nout)
{
    __shared__ alignas(16) unsigned short As[128 * APITCH];  // [m][k]
    __shared__ alignas(16) unsigned short Bs[128 * BPITCH];  // [n][k]

    const int tid  = threadIdx.x;
    const int lane = tid & 31;
    const int wave = tid >> 5;
    const int half = lane >> 4;
    const int lr   = lane & 15;

    const int blockM = blockIdx.y * 128;
    const int blockN = blockIdx.x * 128;
    const int wm = (wave >> 1) * 32;   // 0,32,64,96
    const int wn = (wave & 1) * 64;    // 0,64

    const v8f zero = {0.f,0.f,0.f,0.f,0.f,0.f,0.f,0.f};
    v8f acc[2][4];
#pragma unroll
    for (int i = 0; i < 2; ++i)
#pragma unroll
        for (int j = 0; j < 4; ++j) acc[i][j] = zero;

    for (int k0 = 0; k0 < K; k0 += TILE_K) {
        __syncthreads();
        // Stage A tile 128x32: 8 bf16 (16B) per load/store, fully vectorized
#pragma unroll
        for (int i = 0; i < 2; ++i) {
            int idx = tid + i * 256;              // 0..511
            int m = idx >> 2, kq = (idx & 3) * 8;
            uint4 v = *(const uint4*)(A + (size_t)(blockM + m) * K + k0 + kq);
            *(uint4*)&As[m * APITCH + kq] = v;
        }
        // Stage B tile (pre-transposed weights): same pattern
#pragma unroll
        for (int i = 0; i < 2; ++i) {
            int idx = tid + i * 256;
            int n = idx >> 2, kq = (idx & 3) * 8;
            uint4 v = *(const uint4*)(Wt + (size_t)(blockN + n) * K + k0 + kq);
            *(uint4*)&Bs[n * BPITCH + kq] = v;
        }
        __syncthreads();

        BF16Frag afr[2], bfr[4];
        // A 16x32 bf16: lane half selects +8 within each 16-K group
#pragma unroll
        for (int i = 0; i < 2; ++i) {
            const unsigned short* arow = &As[(wm + i * 16 + lr) * APITCH];
#pragma unroll
            for (int v = 0; v < 8; ++v) {
                int kk = ((v < 4) ? (v * 2) : (v * 2 + 8)) + half * 8;
                afr[i].u[v] = *(const unsigned int*)&arow[kk];
            }
        }
        // B 32x16 bf16: lane half selects +8 within each 16-K group
#pragma unroll
        for (int j = 0; j < 4; ++j) {
            const unsigned short* brow = &Bs[(wn + j * 16 + lr) * BPITCH];
#pragma unroll
            for (int v = 0; v < 8; ++v) {
                int kk = ((v < 4) ? (v * 2) : (v * 2 + 8)) + half * 8;
                bfr[j].u[v] = *(const unsigned int*)&brow[kk];
            }
        }
#pragma unroll
        for (int i = 0; i < 2; ++i)
#pragma unroll
            for (int j = 0; j < 4; ++j)
                acc[i][j] = __builtin_amdgcn_wmma_f32_16x16x32_bf16(
                    false, afr[i].v, false, bfr[j].v,
                    (short)0, acc[i][j], false, false);
    }

    // C/D layout: VGPR r -> row (r + 8*half), col = lane%16
    if (Cbf) {
#pragma unroll
        for (int i = 0; i < 2; ++i)
#pragma unroll
            for (int j = 0; j < 4; ++j) {
                int col = blockN + wn + j * 16 + lr;
#pragma unroll
                for (int r = 0; r < 8; ++r) {
                    int row = blockM + wm + i * 16 + half * 8 + r;
                    if (row < M)
                        Cbf[(size_t)row * Nout + col] = f32_to_bf16(acc[i][j][r]);
                }
            }
    } else {
#pragma unroll
        for (int i = 0; i < 2; ++i)
#pragma unroll
            for (int j = 0; j < 4; ++j) {
                int col = blockN + wn + j * 16 + lr;
                float bv = bias ? bias[col] : 0.0f;
#pragma unroll
                for (int r = 0; r < 8; ++r) {
                    int row = blockM + wm + i * 16 + half * 8 + r;
                    if (row < M)
                        Cf[(size_t)row * Nout + col] = acc[i][j][r] + bv;
                }
            }
    }
}

// ---------------------------------------------------------------------------
// Prep / conversion / aggregation kernels
// ---------------------------------------------------------------------------
__global__ void k_deg_init(float* __restrict__ deg, int n) {
    int i = blockIdx.x * blockDim.x + threadIdx.x;
    if (i < n) deg[i] = 1.0f;   // self loop
}
__global__ void k_deg_count(const int* __restrict__ row, float* __restrict__ deg, int e) {
    int i = blockIdx.x * blockDim.x + threadIdx.x;
    if (i < e) atomicAdd(&deg[row[i]], 1.0f);
}
__global__ void k_dinv(float* __restrict__ d, int n) {
    int i = blockIdx.x * blockDim.x + threadIdx.x;
    if (i < n) d[i] = rsqrtf(d[i]);
}

// fp32 -> bf16, zero-padding rows [n, npad)
__global__ void k_cvt_pad(const float* __restrict__ src, unsigned short* __restrict__ dst,
                          int n, int d, int npad)
{
    long long idx = (long long)blockIdx.x * blockDim.x + threadIdx.x;  // float4 units
    long long total = (long long)npad * d / 4;
    if (idx >= total) return;
    int node = (int)(idx / (d / 4));
    ushort4 o;
    if (node < n) {
        float4 v = ((const float4*)src)[idx];
        o.x = f32_to_bf16(v.x); o.y = f32_to_bf16(v.y);
        o.z = f32_to_bf16(v.z); o.w = f32_to_bf16(v.w);
    } else {
        o.x = o.y = o.z = o.w = 0;
    }
    ((ushort4*)dst)[idx] = o;
}

// relu(h) -> bf16, zero-padding rows [n, npad)
__global__ void k_relu_cvt(const float* __restrict__ h, unsigned short* __restrict__ dst,
                           int n, int d, int npad)
{
    long long idx = (long long)blockIdx.x * blockDim.x + threadIdx.x;
    long long total = (long long)npad * d / 4;
    if (idx >= total) return;
    int node = (int)(idx / (d / 4));
    ushort4 o;
    if (node < n) {
        float4 v = ((const float4*)h)[idx];
        o.x = f32_to_bf16(fmaxf(v.x, 0.f)); o.y = f32_to_bf16(fmaxf(v.y, 0.f));
        o.z = f32_to_bf16(fmaxf(v.z, 0.f)); o.w = f32_to_bf16(fmaxf(v.w, 0.f));
    } else {
        o.x = o.y = o.z = o.w = 0;
    }
    ((ushort4*)dst)[idx] = o;
}

// W[K x Nout] fp32 -> Wt[Nout x K] bf16 (transpose)
__global__ void k_cvt_wT(const float* __restrict__ W, unsigned short* __restrict__ Wt,
                         int K, int Nout)
{
    int idx = blockIdx.x * blockDim.x + threadIdx.x;
    if (idx >= K * Nout) return;
    int k = idx / Nout, n = idx % Nout;       // coalesced read over n
    Wt[(size_t)n * K + k] = f32_to_bf16(W[idx]);
}

// out[i,f] = b[f] + dinv[i]^2 * g[i,f]   (bias + self-loop), g is bf16
__global__ void k_agg_init(const unsigned short* __restrict__ g, const float* __restrict__ bias,
                           const float* __restrict__ dinv, float* __restrict__ out,
                           int n, int d)
{
    long long idx = (long long)blockIdx.x * blockDim.x + threadIdx.x;  // 4-elem units
    long long total = (long long)n * d / 4;
    if (idx >= total) return;
    int node = (int)(idx / (d / 4));
    int f4   = (int)(idx % (d / 4));
    float s = dinv[node]; s *= s;
    ushort4 gq = ((const ushort4*)g)[idx];
    float4 bv = ((const float4*)bias)[f4];
    float4 o;
    o.x = bv.x + s * bf16_to_f32(gq.x); o.y = bv.y + s * bf16_to_f32(gq.y);
    o.z = bv.z + s * bf16_to_f32(gq.z); o.w = bv.w + s * bf16_to_f32(gq.w);
    ((float4*)out)[idx] = o;
}

// One block per edge: out[r,:] += dinv[r]*dinv[c] * g[c,:]  (g bf16, out fp32)
__global__ __launch_bounds__(256) void k_agg_scatter(
    const int* __restrict__ eidx, const float* __restrict__ dinv,
    const unsigned short* __restrict__ g, float* __restrict__ out, int e, int d)
{
    int ed = blockIdx.x;
    int r = eidx[ed];          // destination
    int c = eidx[e + ed];      // source
    float w = dinv[r] * dinv[c];
    const unsigned short* gs = g + (size_t)c * d;
    float*                od = out + (size_t)r * d;
    int f = threadIdx.x * 2;                  // d=512, 256 threads, 2 feats each
    ushort2 gv = *(const ushort2*)&gs[f];
    atomicAdd(&od[f],     w * bf16_to_f32(gv.x));
    atomicAdd(&od[f + 1], w * bf16_to_f32(gv.y));
}

// ---------------------------------------------------------------------------
extern "C" void kernel_launch(void* const* d_in, const int* in_sizes, int n_in,
                              void* d_out, int out_size, void* d_ws, size_t ws_size,
                              hipStream_t stream)
{
    const float* x   = (const float*)d_in[0];
    const int*   ei  = (const int*)d_in[1];
    const float* W1  = (const float*)d_in[2];
    const float* b1  = (const float*)d_in[3];
    const float* W2  = (const float*)d_in[4];
    const float* b2  = (const float*)d_in[5];
    const float* W3  = (const float*)d_in[6];
    const float* b3  = (const float*)d_in[7];
    const float* Wfc = (const float*)d_in[8];
    const float* bfc = (const float*)d_in[9];
    float* out = (float*)d_out;

    const int Din = 256, Dh = 512;
    const int N = in_sizes[0] / Din;                 // 100000
    const int E = in_sizes[1] / 2;                   // 800000
    const int Npad = ((N + 127) / 128) * 128;        // 100096

    // Workspace: dinv | h(fp32 N*Dh) | g(bf16 N*Dh) | abf(bf16 Npad*Dh) | wt(bf16 Dh*Dh)
    char* ws = (char*)d_ws;
    float*          dinv = (float*)ws;                 ws += ((size_t)N * 4 + 1023) & ~(size_t)1023;
    float*          hbuf = (float*)ws;                 ws += (size_t)N * Dh * 4;
    unsigned short* gbf  = (unsigned short*)ws;        ws += (size_t)N * Dh * 2;
    unsigned short* abf  = (unsigned short*)ws;        ws += (size_t)Npad * Dh * 2;
    unsigned short* wt   = (unsigned short*)ws;

    const int T = 256;
    // degrees -> dinv
    k_deg_init <<<(N + T - 1) / T, T, 0, stream>>>(dinv, N);
    k_deg_count<<<(E + T - 1) / T, T, 0, stream>>>(ei, dinv, E);
    k_dinv     <<<(N + T - 1) / T, T, 0, stream>>>(dinv, N);

    long long q4h  = (long long)N * Dh / 4;                 // N*Dh in 4-elem units
    long long p4in = (long long)Npad * Din / 4;
    long long p4h  = (long long)Npad * Dh / 4;
    int gAgg  = (int)((q4h + T - 1) / T);
    int gPin  = (int)((p4in + T - 1) / T);
    int gPh   = (int)((p4h + T - 1) / T);
    dim3 gH(Dh / 128, Npad / 128);
    dim3 gO(Din / 128, Npad / 128);

    // x -> bf16 padded
    k_cvt_pad<<<gPin, T, 0, stream>>>(x, abf, N, Din, Npad);

    // ---- layer 1 ----
    k_cvt_wT      <<<(Din * Dh + T - 1) / T, T, 0, stream>>>(W1, wt, Din, Dh);
    gemm_wmma_bf16<<<gH, T, 0, stream>>>(abf, wt, nullptr, gbf, nullptr, N, Din, Dh);
    k_agg_init    <<<gAgg, T, 0, stream>>>(gbf, b1, dinv, hbuf, N, Dh);
    k_agg_scatter <<<E, T, 0, stream>>>(ei, dinv, gbf, hbuf, E, Dh);
    k_relu_cvt    <<<gPh, T, 0, stream>>>(hbuf, abf, N, Dh, Npad);

    // ---- layer 2 ----
    k_cvt_wT      <<<(Dh * Dh + T - 1) / T, T, 0, stream>>>(W2, wt, Dh, Dh);
    gemm_wmma_bf16<<<gH, T, 0, stream>>>(abf, wt, nullptr, gbf, nullptr, N, Dh, Dh);
    k_agg_init    <<<gAgg, T, 0, stream>>>(gbf, b2, dinv, hbuf, N, Dh);
    k_agg_scatter <<<E, T, 0, stream>>>(ei, dinv, gbf, hbuf, E, Dh);
    k_relu_cvt    <<<gPh, T, 0, stream>>>(hbuf, abf, N, Dh, Npad);

    // ---- layer 3 ----
    k_cvt_wT      <<<(Dh * Dh + T - 1) / T, T, 0, stream>>>(W3, wt, Dh, Dh);
    gemm_wmma_bf16<<<gH, T, 0, stream>>>(abf, wt, nullptr, gbf, nullptr, N, Dh, Dh);
    k_agg_init    <<<gAgg, T, 0, stream>>>(gbf, b3, dinv, hbuf, N, Dh);
    k_agg_scatter <<<E, T, 0, stream>>>(ei, dinv, gbf, hbuf, E, Dh);
    k_relu_cvt    <<<gPh, T, 0, stream>>>(hbuf, abf, N, Dh, Npad);

    // ---- final FC: out = h3 @ Wfc + bfc (fp32 out) ----
    k_cvt_wT      <<<(Dh * Din + T - 1) / T, T, 0, stream>>>(Wfc, wt, Dh, Din);
    gemm_wmma_bf16<<<gO, T, 0, stream>>>(abf, wt, bfc, nullptr, out, N, Dh, Din);
}
